// GNN3_15015205667095
// MI455X (gfx1250) — matrix operand order repaired and checked by reading, not compiled, not run on hardware
//
#include <hip/hip_runtime.h>
#include <math.h>

// ---------------------------------------------------------------------------
// GIN forward for MI455X (gfx1250).
// GEMMs: v_wmma_f32_16x16x32_bf16, weights pre-packed to fragment order
// (1x contiguous 32B load/lane/fragment), A tile staged in LDS as bf16
// fragments (ds_load_b128), 2 M-tiles x 4 N-tiles per wave (8 WMMAs/k-step),
// with double-buffered prefetch of next k-step's A/B fragments.
// ---------------------------------------------------------------------------

typedef __attribute__((ext_vector_type(16))) __bf16 v16bf;
typedef __attribute__((ext_vector_type(8)))  float  v8f;

#define N_NODES  50000
#define N_EDGES  400000
#define C_IN     256
#define HID      512
#define C_OUT    128
#define N_GRAPHS 64
#define M_PAD    50016          // 1563 * 32 : padded node-row count

// ---------------- helper kernels -------------------------------------------

__global__ void copy_f32x4(const float4* __restrict__ in, float4* __restrict__ out, int n4) {
    int i = blockIdx.x * blockDim.x + threadIdx.x;
    if (i < n4) out[i] = in[i];
}

__global__ void zero_f32(float* __restrict__ p, int n) {
    int i = blockIdx.x * blockDim.x + threadIdx.x;
    if (i < n) p[i] = 0.0f;
}

// agg[dst] += x[src], 512 channels per edge. One block per edge.
__global__ void scatter_add_kernel(const float* __restrict__ x, float* __restrict__ agg,
                                   const long long* __restrict__ src,
                                   const long long* __restrict__ dst) {
    int e = blockIdx.x;
    int s = (int)src[e];
    int d = (int)dst[e];
    const float* xs = x + (size_t)s * HID;
    float* ad = agg + (size_t)d * HID;
    for (int c = threadIdx.x; c < HID; c += blockDim.x)
        atomicAdd(ad + c, xs[c]);
}

// segment-sum of node features into per-graph sums + counts.
__global__ void pool_accum(const float* __restrict__ x, const long long* __restrict__ batch,
                           float* __restrict__ sums, float* __restrict__ counts) {
    int nidx = blockIdx.x;
    int g = (int)batch[nidx];
    const float* xr = x + (size_t)nidx * HID;
    float* sr = sums + (size_t)g * HID;
    for (int c = threadIdx.x; c < HID; c += blockDim.x)
        atomicAdd(sr + c, xr[c]);
    if (threadIdx.x == 0) atomicAdd(counts + g, 1.0f);
}

__global__ void pool_finalize(const float* __restrict__ sums, const float* __restrict__ counts,
                              float* __restrict__ pooled) {
    int i = blockIdx.x * blockDim.x + threadIdx.x;   // N_GRAPHS*HID threads
    int g = i / HID;
    pooled[i] = sums[i] / fmaxf(counts[g], 1.0f);
}

// ---------------- weight packing -------------------------------------------
// Pack fp32 row-major B[K,N] into bf16 WMMA-B fragment order:
//   P[((kb*(N/16) + nt)*32 + lane)*16 + e] = B[kb*32 + (lane>>4)*8 + (e>=8?16:0) + (e&7)][nt*16 + (lane&15)]
template <int K, int N>
__global__ void pack_b_kernel(const float* __restrict__ B, __bf16* __restrict__ P) {
    int i = blockIdx.x * blockDim.x + threadIdx.x;   // (K/32)*(N/16)*32 threads
    constexpr int NT = N / 16;
    int lane  = i & 31;
    int entry = i >> 5;
    int nt = entry % NT;
    int kb = entry / NT;
    if (kb >= K / 32) return;
    int n     = nt * 16 + (lane & 15);
    int kbase = kb * 32 + ((lane >> 4) << 3);
    __bf16* dst = P + (size_t)i * 16;
#pragma unroll
    for (int e = 0; e < 16; ++e)
        dst[e] = (__bf16)B[(size_t)(kbase + (e & 7) + ((e >> 3) << 4)) * N + n];
}

// ---------------- fp32x16 -> A fragment conversion --------------------------
__device__ __forceinline__ v16bf cvt_a_frag(const float* __restrict__ p) {
    float4 f0 = *(const float4*)(p);
    float4 f1 = *(const float4*)(p + 4);
    float4 f2 = *(const float4*)(p + 16);
    float4 f3 = *(const float4*)(p + 20);
    v16bf a;
    a[0]  = (__bf16)f0.x; a[1]  = (__bf16)f0.y; a[2]  = (__bf16)f0.z; a[3]  = (__bf16)f0.w;
    a[4]  = (__bf16)f1.x; a[5]  = (__bf16)f1.y; a[6]  = (__bf16)f1.z; a[7]  = (__bf16)f1.w;
    a[8]  = (__bf16)f2.x; a[9]  = (__bf16)f2.y; a[10] = (__bf16)f2.z; a[11] = (__bf16)f2.w;
    a[12] = (__bf16)f3.x; a[13] = (__bf16)f3.y; a[14] = (__bf16)f3.z; a[15] = (__bf16)f3.w;
    return a;
}

// ---------------- WMMA GEMM -------------------------------------------------
// C = epilogue(A[M,K] @ B[K,N]); block = MT*16 rows (blockIdx.x), wave = MT*16 x 64.
// A tile staged once per block into LDS in fragment order, then read back as
// contiguous 32B ds loads. A fragment (16-bit A 16x32, ISA 7.12.2):
//   lane m=L&15, koff=(L>>4)*8, elem e -> K=(e>=8?16:0)+koff+(e&7).
// Hot loop double-buffers the next k-step's A (LDS) and B (global) fragments
// so loads are in flight while the current 4*MT WMMAs execute.
// epilogue: y = (acc + bias[n]) * (scale ? scale[n]*scale_mul : 1) + (shift ? shift[n] : 0)
// act: 0 = none, 1 = relu, 2 = exact gelu.
template <int K, int N, int MT>
__global__ void gemm_wmma_bf16(const float* __restrict__ A, const __bf16* __restrict__ Bp,
                               float* __restrict__ C,
                               const float* __restrict__ bias,
                               const float* __restrict__ scale, float scale_mul,
                               const float* __restrict__ shift,
                               int act) {
    constexpr int KB = K / 32;
    constexpr int NT = N / 16;
    __shared__ alignas(32) __bf16 aFrag[MT * KB * 32 * 16];

    const int tid  = threadIdx.x;
    const int lane = tid & 31;
    const int m0   = blockIdx.x * (MT * 16);

    // ---- stage A tile into LDS as bf16 fragments ----
    for (int ent = tid; ent < MT * KB * 32; ent += blockDim.x) {
        int l  = ent & 31;
        int kb = (ent >> 5) % KB;
        int mt = (ent >> 5) / KB;
        const float* p = A + (size_t)(m0 + mt * 16 + (l & 15)) * K + kb * 32 + ((l >> 4) << 3);
        *(v16bf*)(aFrag + (size_t)ent * 16) = cvt_a_frag(p);
    }
    __syncthreads();

    const v16bf* __restrict__ Av = (const v16bf*)aFrag;
    const v16bf* __restrict__ Bv = (const v16bf*)Bp;
    const int nt0 = (tid >> 5) << 2;   // first of 4 N-tiles for this wave

    v8f acc[MT][4];
#pragma unroll
    for (int mt = 0; mt < MT; ++mt)
#pragma unroll
        for (int t = 0; t < 4; ++t)
            acc[mt][t] = (v8f){0.f,0.f,0.f,0.f,0.f,0.f,0.f,0.f};

    // ---- double-buffered k-loop ----
    v16bf a_cur[MT], b_cur[4];
#pragma unroll
    for (int mt = 0; mt < MT; ++mt)
        a_cur[mt] = Av[(mt * KB + 0) * 32 + lane];
#pragma unroll
    for (int t = 0; t < 4; ++t)
        b_cur[t] = Bv[((size_t)0 * NT + (nt0 + t)) * 32 + lane];

#pragma unroll
    for (int kb = 0; kb < KB; ++kb) {
        v16bf a_nxt[MT], b_nxt[4];
        if (kb + 1 < KB) {
#pragma unroll
            for (int t = 0; t < 4; ++t)
                b_nxt[t] = Bv[((size_t)(kb + 1) * NT + (nt0 + t)) * 32 + lane];
#pragma unroll
            for (int mt = 0; mt < MT; ++mt)
                a_nxt[mt] = Av[(mt * KB + (kb + 1)) * 32 + lane];
        }
#pragma unroll
        for (int t = 0; t < 4; ++t)
#pragma unroll
            for (int mt = 0; mt < MT; ++mt)
                acc[mt][t] = __builtin_amdgcn_wmma_f32_16x16x32_bf16(
                    false, a_cur[mt], false, b_cur[t], (short)0, acc[mt][t], false, false);
        if (kb + 1 < KB) {
#pragma unroll
            for (int mt = 0; mt < MT; ++mt) a_cur[mt] = a_nxt[mt];
#pragma unroll
            for (int t = 0; t < 4; ++t) b_cur[t] = b_nxt[t];
        }
    }

    // D layout (32-bit C/D 16x16): elem v -> row 16*tile + (L>>4)*8 + v, col nt*16 + (L&15)
    const int nb   = lane & 15;
    const int mrow = (lane >> 4) << 3;
#pragma unroll
    for (int t = 0; t < 4; ++t) {
        int n = (nt0 + t) * 16 + nb;
        float bi = bias  ? bias[n]              : 0.0f;
        float sc = scale ? scale[n] * scale_mul : 1.0f;
        float sh = shift ? shift[n]             : 0.0f;
#pragma unroll
        for (int mt = 0; mt < MT; ++mt) {
#pragma unroll
            for (int v = 0; v < 8; ++v) {
                int m = m0 + mt * 16 + mrow + v;
                float y = (acc[mt][t][v] + bi) * sc + sh;
                if (act == 1)      y = fmaxf(y, 0.0f);
                else if (act == 2) y = 0.5f * y * (1.0f + erff(y * 0.70710678118654752f));
                C[(size_t)m * N + n] = y;
            }
        }
    }
}

// ---------------- launcher --------------------------------------------------

extern "C" void kernel_launch(void* const* d_in, const int* in_sizes, int n_in,
                              void* d_out, int out_size, void* d_ws, size_t ws_size,
                              hipStream_t stream) {
    const float*     x       = (const float*)d_in[0];
    const long long* ei      = (const long long*)d_in[1];   // int64 [2, N_EDGES]
    const long long* batch   = (const long long*)d_in[2];   // int64 [N_NODES]
    const float*     enc_w   = (const float*)d_in[3];
    const float*     enc_b   = (const float*)d_in[4];
    const float*     conv_w0 = (const float*)d_in[5];       // [2, HID, HID]
    const float*     conv_b0 = (const float*)d_in[6];       // [2, HID]
    const float*     bn0_g   = (const float*)d_in[7];
    const float*     bn0_b   = (const float*)d_in[8];
    const float*     conv_w1 = (const float*)d_in[9];
    const float*     conv_b1 = (const float*)d_in[10];
    const float*     bn1_g   = (const float*)d_in[11];
    const float*     bn1_b   = (const float*)d_in[12];
    const float*     lin_w   = (const float*)d_in[13];
    const float*     lin_b   = (const float*)d_in[14];
    const float*     clf_w   = (const float*)d_in[15];
    const float*     clf_b   = (const float*)d_in[16];
    float*           out     = (float*)d_out;

    // workspace carve-out
    char*  ws  = (char*)d_ws;
    size_t off = 0;
    auto take = [&](size_t bytes) -> char* {
        char* p = ws + off;
        off += (bytes + 255) & ~(size_t)255;
        return p;
    };
    float*  bufX   = (float*)take((size_t)M_PAD * HID * sizeof(float));
    float*  bufAgg = (float*)take((size_t)M_PAD * HID * sizeof(float));
    float*  bufT   = (float*)take((size_t)M_PAD * HID * sizeof(float));
    float*  sums   = (float*)take((size_t)N_GRAPHS * HID * sizeof(float));
    float*  counts = (float*)take((size_t)N_GRAPHS * sizeof(float));
    float*  pooled = (float*)take((size_t)N_GRAPHS * HID * sizeof(float));
    float*  gbuf   = (float*)take((size_t)N_GRAPHS * HID * sizeof(float));
    __bf16* pEnc   = (__bf16*)take((size_t)C_IN * HID * sizeof(__bf16));
    __bf16* pW0a   = (__bf16*)take((size_t)HID * HID * sizeof(__bf16));
    __bf16* pW1a   = (__bf16*)take((size_t)HID * HID * sizeof(__bf16));
    __bf16* pW0b   = (__bf16*)take((size_t)HID * HID * sizeof(__bf16));
    __bf16* pW1b   = (__bf16*)take((size_t)HID * HID * sizeof(__bf16));
    __bf16* pLin   = (__bf16*)take((size_t)HID * HID * sizeof(__bf16));
    __bf16* pClf   = (__bf16*)take((size_t)HID * C_OUT * sizeof(__bf16));

    const float bn_rsqrt = 0.9999950000374997f;   // 1/sqrt(1 + 1e-5)
    const long long* srcE = ei;
    const long long* dstE = ei + N_EDGES;

    // ---- pack all weights into WMMA fragment order (bf16) ----
    {
        int t_enc = (C_IN / 32) * (HID / 16) * 32;    // 8192
        int t_hid = (HID / 32) * (HID / 16) * 32;     // 16384
        int t_clf = (HID / 32) * (C_OUT / 16) * 32;   // 4096
        pack_b_kernel<C_IN, HID><<<(t_enc + 255) / 256, 256, 0, stream>>>(enc_w, pEnc);
        pack_b_kernel<HID, HID><<<(t_hid + 255) / 256, 256, 0, stream>>>(conv_w0, pW0a);
        pack_b_kernel<HID, HID><<<(t_hid + 255) / 256, 256, 0, stream>>>(conv_w0 + (size_t)HID * HID, pW0b);
        pack_b_kernel<HID, HID><<<(t_hid + 255) / 256, 256, 0, stream>>>(conv_w1, pW1a);
        pack_b_kernel<HID, HID><<<(t_hid + 255) / 256, 256, 0, stream>>>(conv_w1 + (size_t)HID * HID, pW1b);
        pack_b_kernel<HID, HID><<<(t_hid + 255) / 256, 256, 0, stream>>>(lin_w, pLin);
        pack_b_kernel<HID, C_OUT><<<(t_clf + 255) / 256, 256, 0, stream>>>(clf_w, pClf);
    }

    // zero pad rows of bufX so padded GEMM tiles are deterministic
    zero_f32<<<((M_PAD - N_NODES) * HID + 255) / 256, 256, 0, stream>>>(
        bufX + (size_t)N_NODES * HID, (M_PAD - N_NODES) * HID);

    // encoder: bufX[0:N_NODES] = x @ enc_w + enc_b   (MT=1: external input, 16-row tiles)
    gemm_wmma_bf16<C_IN, HID, 1><<<N_NODES / 16, (HID / 64) * 32, 0, stream>>>(
        x, pEnc, bufX, enc_b, nullptr, 1.0f, nullptr, 0);

    const __bf16* w0p[2] = { pW0a, pW0b };
    const __bf16* w1p[2] = { pW1a, pW1b };
    for (int layer = 0; layer < 2; ++layer) {
        // agg = x  (so scatter-add yields x + sum(neighbors) directly)
        int n4 = M_PAD * HID / 4;
        copy_f32x4<<<(n4 + 255) / 256, 256, 0, stream>>>((const float4*)bufX, (float4*)bufAgg, n4);
        scatter_add_kernel<<<N_EDGES, 128, 0, stream>>>(bufX, bufAgg, srcE, dstE);

        // t = relu(bn0((x+agg) @ w0 + b0))
        gemm_wmma_bf16<HID, HID, 2><<<M_PAD / 32, (HID / 64) * 32, 0, stream>>>(
            bufAgg, w0p[layer], bufT,
            conv_b0 + layer * HID, bn0_g + layer * HID, bn_rsqrt, bn0_b + layer * HID, 1);

        // x = bn1(t @ w1 + b1), relu only between layers
        gemm_wmma_bf16<HID, HID, 2><<<M_PAD / 32, (HID / 64) * 32, 0, stream>>>(
            bufT, w1p[layer], bufX,
            conv_b1 + layer * HID, bn1_g + layer * HID, bn_rsqrt, bn1_b + layer * HID,
            layer == 0 ? 1 : 0);
    }

    // global mean pool (valid nodes only)
    zero_f32<<<(N_GRAPHS * HID + 255) / 256, 256, 0, stream>>>(sums, N_GRAPHS * HID);
    zero_f32<<<1, 64, 0, stream>>>(counts, N_GRAPHS);
    pool_accum<<<N_NODES, 128, 0, stream>>>(bufX, batch, sums, counts);
    pool_finalize<<<(N_GRAPHS * HID) / 256, 256, 0, stream>>>(sums, counts, pooled);

    // g = gelu(pooled @ lin_w + lin_b)
    gemm_wmma_bf16<HID, HID, 2><<<N_GRAPHS / 32, (HID / 64) * 32, 0, stream>>>(
        pooled, pLin, gbuf, lin_b, nullptr, 1.0f, nullptr, 2);

    // out = g @ clf_w + clf_b
    gemm_wmma_bf16<HID, C_OUT, 2><<<N_GRAPHS / 32, (C_OUT / 64) * 32, 0, stream>>>(
        gbuf, pClf, out, clf_b, nullptr, 1.0f, nullptr, 0);

    (void)in_sizes; (void)n_in; (void)out_size; (void)ws_size;
}
